// ActorCriticV3_60498909331525
// MI455X (gfx1250) — compile-verified, matrix operand order checked by reading, not run on hardware
//
#include <hip/hip_runtime.h>

// ---------------- constants ----------------
#define BATCH 512
#define TT 162
#define TPAD 176                 // 11 * 16
#define MR (BATCH*TPAD)          // 90112 padded token rows
#define DD 128
#define OBSROW 16215             // T*RPT + GDIM

typedef unsigned short u16;
typedef unsigned int   u32;
typedef __attribute__((ext_vector_type(16))) __bf16 bf16x16;
typedef __attribute__((ext_vector_type(8)))  float  f32x8;

union FragA { bf16x16 v; u32 dw[8]; };

__device__ __forceinline__ u16 f2bf(float f){
  u32 u = __float_as_uint(f);
  u32 r = u + 0x7FFFu + ((u >> 16) & 1u);
  return (u16)(r >> 16);
}
__device__ __forceinline__ float bf2f(u16 h){ return __uint_as_float(((u32)h) << 16); }
// bf16 16x16x32 A/B fragment K pattern: vgpr v, lane-half lh -> k base
__device__ __forceinline__ int kpat(int v, int lh){ return ((v>>2)<<4) + (lh<<3) + ((v&3)<<1); }

__device__ __forceinline__ f32x8 wmma_bf(const FragA& a, const FragA& b, f32x8 c){
  return __builtin_amdgcn_wmma_f32_16x16x32_bf16(false, a.v, false, b.v, (short)0, c, false, false);
}

// ---------------- generic bf16 WMMA GEMM: D = act(A @ W^T + bias) ----------------
// A: (M, Kpad) bf16 row-major.  W: (Npad, Kpad) bf16 row-major (so B[k][n] = W[n][k]).
// act: 0 none, 1 relu, 2 gelu(exact), 3 sigmoid.
// outF (fp32, ld=ldoF) / outB (bf16, ld=ldoB) optional; outHead scatters bf16 into (B,H,TPAD,32).
// Double-buffered K loop: next K tile's global loads issue under current WMMAs.
template<int WM, int WN>
__global__ __launch_bounds__(64) void gemm_k(const u16* __restrict__ A, int lda,
    const u16* __restrict__ W, int ldw,
    const float* __restrict__ bias, int biasN,
    int M, int Npad, int Kpad, int mtiles, int ntiles, int act,
    float* outF, int ldoF, u16* outB, int ldoB, u16* outHead)
{
  int w = threadIdx.x >> 5, lane = threadIdx.x & 31;
  int gw = blockIdx.x * 2 + w;
  if (gw >= mtiles * ntiles) return;
  int tm = gw % mtiles, tn = gw / mtiles;
  int m0 = tm * 16 * WM, n0 = tn * 16 * WN;
  int l15 = lane & 15, lh = lane >> 4;

  const u16* arow[WM]; const u16* wrow[WN];
#pragma unroll
  for (int i = 0; i < WM; ++i) arow[i] = A + (long)(m0 + i*16 + l15) * lda;
#pragma unroll
  for (int j = 0; j < WN; ++j) wrow[j] = W + (long)(n0 + j*16 + l15) * ldw;

  f32x8 acc[WM][WN];
  f32x8 zero = {0.f,0.f,0.f,0.f,0.f,0.f,0.f,0.f};
#pragma unroll
  for (int i = 0; i < WM; ++i)
#pragma unroll
    for (int j = 0; j < WN; ++j) acc[i][j] = zero;

  FragA a0[WM], b0[WN], a1[WM], b1[WN];
  auto load0 = [&](int k0){
#pragma unroll
    for (int i = 0; i < WM; ++i)
#pragma unroll
      for (int v = 0; v < 8; ++v)
        a0[i].dw[v] = *(const u32*)(arow[i] + k0 + kpat(v, lh));
#pragma unroll
    for (int j = 0; j < WN; ++j)
#pragma unroll
      for (int v = 0; v < 8; ++v)
        b0[j].dw[v] = *(const u32*)(wrow[j] + k0 + kpat(v, lh));
    __builtin_prefetch(arow[0] + k0 + 64, 0, 0);   // gfx1250 global_prefetch_b8
  };
  auto load1 = [&](int k0){
#pragma unroll
    for (int i = 0; i < WM; ++i)
#pragma unroll
      for (int v = 0; v < 8; ++v)
        a1[i].dw[v] = *(const u32*)(arow[i] + k0 + kpat(v, lh));
#pragma unroll
    for (int j = 0; j < WN; ++j)
#pragma unroll
      for (int v = 0; v < 8; ++v)
        b1[j].dw[v] = *(const u32*)(wrow[j] + k0 + kpat(v, lh));
    __builtin_prefetch(wrow[0] + k0 + 64, 0, 0);
  };
  auto mma0 = [&]{
#pragma unroll
    for (int i = 0; i < WM; ++i)
#pragma unroll
      for (int j = 0; j < WN; ++j) acc[i][j] = wmma_bf(a0[i], b0[j], acc[i][j]);
  };
  auto mma1 = [&]{
#pragma unroll
    for (int i = 0; i < WM; ++i)
#pragma unroll
      for (int j = 0; j < WN; ++j) acc[i][j] = wmma_bf(a1[i], b1[j], acc[i][j]);
  };

  load0(0);
  for (int k0 = 0; ; ) {
    bool more = (k0 + 32 < Kpad);
    if (more) load1(k0 + 32);
    mma0();
    if (!more) break;
    k0 += 32;
    more = (k0 + 32 < Kpad);
    if (more) load0(k0 + 32);
    mma1();
    if (!more) break;
    k0 += 32;
  }

#pragma unroll
  for (int i = 0; i < WM; ++i) {
#pragma unroll
    for (int j = 0; j < WN; ++j) {
      int n = n0 + j*16 + l15;
      float bs = (bias && n < biasN) ? bias[n] : 0.f;
#pragma unroll
      for (int r = 0; r < 8; ++r) {
        int m = m0 + i*16 + r + lh*8;
        float vv = acc[i][j][r] + bs;
        if (act == 1) vv = fmaxf(vv, 0.f);
        else if (act == 2) vv = 0.5f * vv * (1.f + erff(vv * 0.70710678f));
        else if (act == 3) vv = 1.f / (1.f + __expf(-vv));
        if (outF) outF[(long)m * ldoF + n] = vv;
        if (outB) outB[(long)m * ldoB + n] = f2bf(vv);
        if (outHead) {
          int bb = m / TPAD, t = m - bb * TPAD;
          int hh = n >> 5, dd = n & 31;
          outHead[(((long)(bb*4 + hh)) * TPAD + t) * 32 + dd] = f2bf(vv);
        }
      }
    }
  }
}

// ---------------- fused masked MHA per (b,h) ----------------
__global__ __launch_bounds__(64) void attn_k(const u16* __restrict__ q,
    const u16* __restrict__ k, const u16* __restrict__ v,
    const u32* __restrict__ mb, u16* __restrict__ ob)
{
  __shared__ u16 Ks[TPAD*32];
  __shared__ u16 VsT[32*TPAD];
  __shared__ u32 Mb[TPAD*8];
  __shared__ float Sb[2][16*TPAD];
  __shared__ u16 Pb[2][16*TPAD];
  int bh = blockIdx.x;
  const u16* kg = k + (long)bh * TPAD * 32;
  const u16* vg = v + (long)bh * TPAD * 32;
  for (int i = threadIdx.x; i < TPAD*32; i += 64) Ks[i] = kg[i];
  for (int i = threadIdx.x; i < TPAD*32; i += 64) { int t = i >> 5, n = i & 31; VsT[n*TPAD + t] = vg[i]; }
  for (int i = threadIdx.x; i < TPAD*8;  i += 64) Mb[i] = mb[i];
  __syncthreads();
  int w = threadIdx.x >> 5, lane = threadIdx.x & 31, l15 = lane & 15, lh = lane >> 4;
  float* S = Sb[w]; u16* P = Pb[w];
  const u16* qg = q + (long)bh * TPAD * 32;
  const float scale = 0.17677669529663687f;   // 1/sqrt(32)
  f32x8 zero = {0.f,0.f,0.f,0.f,0.f,0.f,0.f,0.f};
  for (int s = w; s < 11; s += 2) {
    int t0 = s * 16;
    FragA a;
#pragma unroll
    for (int vv = 0; vv < 8; ++vv) a.dw[vv] = *(const u32*)(qg + (t0 + l15)*32 + kpat(vv, lh));
    for (int j = 0; j < 11; ++j) {
      FragA bf;
#pragma unroll
      for (int vv = 0; vv < 8; ++vv) bf.dw[vv] = *(const u32*)(&Ks[(j*16 + l15)*32 + kpat(vv, lh)]);
      f32x8 c = zero;
      c = wmma_bf(a, bf, c);
      int tc = j*16 + l15;
      int wrd = tc >> 5, sh = tc & 31;
#pragma unroll
      for (int r = 0; r < 8; ++r) {
        int m = r + lh*8, t = t0 + m;
        // Mb rows >= TT are all-zero, so no t<TT test needed: branchless select
        u32 word = Mb[t*8 + wrd];
        float val = ((word >> sh) & 1u) ? c[r]*scale : -1e30f;
        S[m*TPAD + tc] = val;
      }
    }
    // parallel softmax: row = lane&15 handled by lane pair (lane, lane^16)
    {
      int r0 = l15;
      float* row = &S[r0 * TPAD];
      u16*  pr  = &P[r0 * TPAD];
      int c0 = lh * 88;
      int cend = c0 + 88; if (cend > TT) cend = TT;
      float mx = -1e30f;
      for (int t = c0; t < cend; ++t) mx = fmaxf(mx, row[t]);
      mx = fmaxf(mx, __shfl_xor(mx, 16));
      float sum = 0.f;
      for (int t = c0; t < cend; ++t) { float e = __expf(row[t]-mx); row[t] = e; sum += e; }
      sum += __shfl_xor(sum, 16);
      float inv = 1.f / sum;
      for (int t = c0; t < cend; ++t) pr[t] = f2bf(row[t]*inv);
      if (lh) for (int t = TT; t < TPAD; ++t) pr[t] = 0;   // zero pad cols (K dim of P@V)
    }
    f32x8 o0 = zero, o1 = zero;
    for (int kk = 0; kk < 11; ++kk) {
      FragA pa, v0, v1;
#pragma unroll
      for (int vv = 0; vv < 8; ++vv) {
        pa.dw[vv] = *(const u32*)(&P[l15*TPAD + kk*16 + kpat(vv, lh)]);
        v0.dw[vv] = *(const u32*)(&VsT[(l15)*TPAD + kk*16 + kpat(vv, lh)]);
        v1.dw[vv] = *(const u32*)(&VsT[(16 + l15)*TPAD + kk*16 + kpat(vv, lh)]);
      }
      o0 = wmma_bf(pa, v0, o0);
      o1 = wmma_bf(pa, v1, o1);
    }
    int bb = bh >> 2, hh = bh & 3;
#pragma unroll
    for (int r = 0; r < 8; ++r) {
      int t = t0 + r + lh*8;
      if (t < TT) {
        long ro = ((long)(bb*TPAD + t)) * 128 + hh*32;
        ob[ro + l15]      = f2bf(o0[r]);
        ob[ro + 16 + l15] = f2bf(o1[r]);
      }
    }
  }
}

// ---------------- LayerNorm variants (one wave per 128-col row) ----------------
// mode 0: x = relu(LN(src)) + ge[b]       rows = B*T
// mode 1: x = LN(x + src)                 rows = B*T
// mode 2: ctx[b,j*128+..] = LN(src row b*4+j)   rows = B*3
__global__ __launch_bounds__(64) void ln_k(int mode, const float* __restrict__ src,
    float* x, u16* xb, const float* __restrict__ g, const float* __restrict__ be,
    const float* __restrict__ ge, float* ctx, int rows)
{
  int row = blockIdx.x*2 + (threadIdx.x >> 5);
  if (row >= rows) return;
  int lane = threadIdx.x & 31;
  long base; int brow, bidx = 0;
  if (mode == 2) { int b = row/3, j = row - 3*b; brow = b; base = ((long)(b*4 + j))*128; bidx = b*512 + j*128; }
  else           { int b = row/TT, t = row - TT*b; brow = b; base = ((long)b*TPAD + t)*128; }
  float v[4];
#pragma unroll
  for (int i = 0; i < 4; ++i) {
    int col = lane + 32*i;
    float s = src[base + col];
    if (mode == 1) s += x[base + col];
    v[i] = s;
  }
  float s0 = v[0]+v[1]+v[2]+v[3];
  for (int o = 16; o > 0; o >>= 1) s0 += __shfl_xor(s0, o);
  float mu = s0 * (1.f/128.f);
  float q0 = 0.f;
#pragma unroll
  for (int i = 0; i < 4; ++i) { float d = v[i]-mu; q0 += d*d; }
  for (int o = 16; o > 0; o >>= 1) q0 += __shfl_xor(q0, o);
  float inv = rsqrtf(q0 * (1.f/128.f) + 1e-5f);
#pragma unroll
  for (int i = 0; i < 4; ++i) {
    int col = lane + 32*i;
    float y = (v[i]-mu)*inv*g[col] + be[col];
    if (mode == 2) { ctx[bidx + col] = y; }
    else {
      if (mode == 0) y = fmaxf(y, 0.f) + ge[brow*128 + col];
      x[base + col] = y;
      xb[base + col] = f2bf(y);
    }
  }
}

// ---------------- small helper kernels ----------------
__global__ void convW(const float* __restrict__ src, u16* __restrict__ dst, int N, int K, int Np, int Kp){
  long i = (long)blockIdx.x*256 + threadIdx.x;
  if (i >= (long)Np*Kp) return;
  int n = (int)(i / Kp), kk = (int)(i - (long)n*Kp);
  float v = (n < N && kk < K) ? src[(long)n*K + kk] : 0.f;
  dst[i] = f2bf(v);
}
__global__ void prep_terr(const float* __restrict__ obs, u16* __restrict__ At){
  long i = (long)blockIdx.x*256 + threadIdx.x;
  if (i >= (long)MR*128) return;
  int m = (int)(i >> 7), kk = (int)(i & 127);
  int b = m / TPAD, t = m - b*TPAD;
  float v = (t < TT && kk < 100) ? obs[(long)b*OBSROW + t*100 + kk] : 0.f;
  At[i] = f2bf(v);
}
__global__ void prep_g(const float* __restrict__ obs, u16* __restrict__ Ag){
  int i = blockIdx.x*256 + threadIdx.x;
  if (i >= BATCH*32) return;
  int b = i >> 5, kk = i & 31;
  float v = (kk < 15) ? obs[(long)b*OBSROW + TT*100 + kk] : 0.f;
  Ag[i] = f2bf(v);
}
__global__ void mask_k(const unsigned char* __restrict__ adj, u32* __restrict__ mb){
  int i = blockIdx.x*256 + threadIdx.x;
  if (i >= TPAD*8) return;
  int t = i >> 3, w = i & 7;
  u32 word = 0;
  if (t < TT)
    for (int bit = 0; bit < 32; ++bit) {
      int tc = w*32 + bit;
      if (tc < TT && (tc == t || adj[t*TT + tc] != 0)) word |= 1u << bit;
    }
  mb[i] = word;
}
__global__ void qp_k(const float* __restrict__ tok, const float* __restrict__ inW,
                     const float* __restrict__ inb, float* __restrict__ qp){
  int i = blockIdx.x*128 + threadIdx.x;
  if (i >= 384) return;
  int j = i >> 7, n = i & 127;
  float s = inb[n];
  for (int kk = 0; kk < 128; ++kk) s += tok[j*128 + kk] * inW[n*128 + kk];
  qp[i] = s;
}
__global__ __launch_bounds__(128) void pool_attn(const float* __restrict__ qp,
    const u16* __restrict__ kp, const u16* __restrict__ vp, u16* __restrict__ att3)
{
  int b = blockIdx.x;
  __shared__ float sc[TT];
  __shared__ float qv[32];
  const float scale = 0.17677669529663687f;
  for (int h = 0; h < 4; ++h) {
    const u16* kh = kp + ((long)(b*4 + h))*TPAD*32;
    const u16* vh = vp + ((long)(b*4 + h))*TPAD*32;
    for (int j = 0; j < 3; ++j) {
      if (threadIdx.x < 32) qv[threadIdx.x] = qp[j*128 + h*32 + threadIdx.x];
      __syncthreads();
      for (int t = threadIdx.x; t < TT; t += 128) {
        float s = 0.f;
        for (int d = 0; d < 32; ++d) s += qv[d] * bf2f(kh[t*32 + d]);
        sc[t] = s * scale;
      }
      __syncthreads();
      if (threadIdx.x == 0) {
        float mx = -1e30f;
        for (int t = 0; t < TT; ++t) mx = fmaxf(mx, sc[t]);
        float sum = 0.f;
        for (int t = 0; t < TT; ++t) { float e = __expf(sc[t]-mx); sc[t] = e; sum += e; }
        float inv = 1.f/sum;
        for (int t = 0; t < TT; ++t) sc[t] *= inv;
      }
      __syncthreads();
      if (threadIdx.x < 32) {
        int n = threadIdx.x;
        float o = 0.f;
        for (int t = 0; t < TT; ++t) o += sc[t] * bf2f(vh[t*32 + n]);
        att3[((long)(b*4 + j))*128 + h*32 + n] = f2bf(o);
      }
      __syncthreads();
    }
  }
}
__global__ void ctx_build(float* __restrict__ ctx, const float* __restrict__ ge, u16* __restrict__ ctxb){
  int i = blockIdx.x*256 + threadIdx.x;
  if (i >= BATCH*512) return;
  int b = i >> 9, c = i & 511;
  float v;
  if (c < 384) v = ctx[i];
  else { v = ge[b*128 + (c - 384)]; ctx[i] = v; }
  ctxb[i] = f2bf(v);
}
__global__ void gather_out(const float* __restrict__ puro, const float* __restrict__ atko,
                           const float* __restrict__ rnfo, const float* __restrict__ vo,
                           float* __restrict__ out){
  long i = (long)blockIdx.x*256 + threadIdx.x;
  long n_pur = (long)BATCH*13, n_tok = (long)BATCH*TT;
  if (i < n_pur) { int b = (int)(i/13), n = (int)(i%13); out[(long)b*337 + n] = puro[b*16 + n]; return; }
  i -= n_pur;
  if (i < n_tok) { int b = (int)(i/TT), t = (int)(i%TT); out[(long)b*337 + 13 + t] = atko[((long)b*TPAD + t)*16]; return; }
  i -= n_tok;
  if (i < n_tok) { int b = (int)(i/TT), t = (int)(i%TT); out[(long)b*337 + 175 + t] = rnfo[((long)b*TPAD + t)*16]; return; }
  i -= n_tok;
  if (i < BATCH) { out[(long)BATCH*337 + i] = vo[i*16]; }
}

// ---------------- host side ----------------
static void rungemm(hipStream_t st, int wm, int wn, const u16* A, int lda, const u16* W, int ldw,
    const float* bias, int biasN, int M, int Np, int Kp, int act,
    float* oF, int ldoF, u16* oB, int ldoB, u16* oH)
{
  int mt = M / (16*wm), nt = Np / (16*wn);
  int waves = mt*nt, blocks = (waves + 1) / 2;
  if (wm == 4 && wn == 4)
    gemm_k<4,4><<<blocks,64,0,st>>>(A,lda,W,ldw,bias,biasN,M,Np,Kp,mt,nt,act,oF,ldoF,oB,ldoB,oH);
  else
    gemm_k<4,1><<<blocks,64,0,st>>>(A,lda,W,ldw,bias,biasN,M,Np,Kp,mt,nt,act,oF,ldoF,oB,ldoB,oH);
}

// Input index map: assumes JAX pytree flatten (dict keys sorted alphabetically).
enum {
  IN_ADJ=0, IN_OBS=1,
  IN_ATK1_W=2, IN_ATK1_B=3, IN_ATK2_W=4, IN_ATK2_B=5,
  IN_CMP_W=6, IN_CMP_B=7, IN_EMB_W=8, IN_EMB_B=9,
  IN_EMB_LN_B=10, IN_EMB_LN_G=11,
  IN_G1_W=12, IN_G1_B=13, IN_G2_W=14, IN_G2_B=15,
  IN_GNN0=16, // per layer (stride 16): ff1W,ff1b,ff2W,ff2b,kW,kb,ln1b,ln1g,ln2b,ln2g,outW,outb,qW,qb,vW,vb
  IN_POOL_INW=64, IN_POOL_INB=65, IN_POOL_LNB=66, IN_POOL_LNG=67,
  IN_POOL_OPW=68, IN_POOL_OPB=69, IN_POOL_TOK=70,
  IN_PUR1_W=71, IN_PUR1_B=72, IN_PUR2_W=73, IN_PUR2_B=74,
  IN_RNF1_W=75, IN_RNF1_B=76, IN_RNF2_W=77, IN_RNF2_B=78,
  IN_V1_W=79, IN_V1_B=80, IN_V2_W=81, IN_V2_B=82, IN_V3_W=83, IN_V3_B=84,
};

extern "C" void kernel_launch(void* const* d_in, const int* in_sizes, int n_in,
                              void* d_out, int out_size, void* d_ws, size_t ws_size,
                              hipStream_t stream) {
  char* W8 = (char*)d_ws;
  size_t off = 0;
  auto alloc = [&](size_t bytes) -> void* { void* p = W8 + off; off += (bytes + 255) & ~(size_t)255; return p; };
  auto F = [&](int i) -> const float* { return (const float*)d_in[i]; };

  u16*  At   = (u16*) alloc((size_t)MR*128*2);
  u16*  Ag   = (u16*) alloc((size_t)BATCH*32*2);
  u16*  Cc   = (u16*) alloc((size_t)MR*64*2);
  float* tmp = (float*)alloc((size_t)MR*128*4);
  float* x   = (float*)alloc((size_t)MR*128*4);
  u16*  xb   = (u16*) alloc((size_t)MR*128*2);
  u16*  qb   = (u16*) alloc((size_t)MR*128*2);
  u16*  kb   = (u16*) alloc((size_t)MR*128*2);
  u16*  vb   = (u16*) alloc((size_t)MR*128*2);
  u16*  ob   = (u16*) alloc((size_t)MR*128*2);
  u16*  f1   = (u16*) alloc((size_t)MR*256*2);
  u16*  h1   = (u16*) alloc((size_t)BATCH*64*2);
  float* ge  = (float*)alloc((size_t)BATCH*128*4);
  u32*  mbits= (u32*) alloc((size_t)TPAD*8*4);
  float* qp  = (float*)alloc(384*4);
  u16*  att3 = (u16*) alloc((size_t)2048*128*2);
  float* proj3=(float*)alloc((size_t)2048*128*4);
  float* ctx = (float*)alloc((size_t)BATCH*512*4);
  u16*  ctxb = (u16*) alloc((size_t)BATCH*512*2);
  u16*  p1   = (u16*) alloc((size_t)BATCH*128*2);
  float* puro= (float*)alloc((size_t)BATCH*16*4);
  u16*  a1   = (u16*) alloc((size_t)MR*64*2);
  float* atko= (float*)alloc((size_t)MR*16*4);
  u16*  r1   = (u16*) alloc((size_t)MR*64*2);
  float* rnfo= (float*)alloc((size_t)MR*16*4);
  u16*  vh1  = (u16*) alloc((size_t)BATCH*128*2);
  u16*  vh2  = (u16*) alloc((size_t)BATCH*64*2);
  float* vo  = (float*)alloc((size_t)BATCH*16*4);
  // bf16 weights (padded: N->mult16, K->mult32)
  u16* wg1 = (u16*)alloc(64*32*2);    u16* wg2 = (u16*)alloc(128*64*2);
  u16* wcmp= (u16*)alloc(48*128*2);   u16* wemb= (u16*)alloc(128*64*2);
  u16 *wq[3], *wk[3], *wv[3], *wo[3], *wf1[3], *wf2[3];
  for (int l = 0; l < 3; ++l) {
    wq[l]=(u16*)alloc(128*128*2); wk[l]=(u16*)alloc(128*128*2); wv[l]=(u16*)alloc(128*128*2);
    wo[l]=(u16*)alloc(128*128*2); wf1[l]=(u16*)alloc(256*128*2); wf2[l]=(u16*)alloc(128*256*2);
  }
  u16* wpk = (u16*)alloc(128*128*2); u16* wpv = (u16*)alloc(128*128*2); u16* wpo = (u16*)alloc(128*128*2);
  u16* wpur1=(u16*)alloc(128*512*2); u16* wpur2=(u16*)alloc(16*128*2);
  u16* watk1=(u16*)alloc(64*128*2);  u16* watk2=(u16*)alloc(16*64*2);
  u16* wrnf1=(u16*)alloc(64*128*2);  u16* wrnf2=(u16*)alloc(16*64*2);
  u16* wv1 = (u16*)alloc(128*512*2); u16* wv2 = (u16*)alloc(64*128*2); u16* wv3 = (u16*)alloc(16*64*2);

  if (off > ws_size) return;   // workspace insufficient; bail deterministically
  hipMemsetAsync(d_ws, 0, off, stream);

  auto conv = [&](int idx, long src_off, int N, int K, int Np, int Kp, u16* dst){
    long tot = (long)Np*Kp;
    convW<<<(unsigned)((tot + 255)/256), 256, 0, stream>>>(F(idx) + src_off, dst, N, K, Np, Kp);
  };
  conv(IN_G1_W, 0, 64, 15, 64, 32, wg1);
  conv(IN_G2_W, 0, 128, 64, 128, 64, wg2);
  conv(IN_CMP_W, 0, 37, 100, 48, 128, wcmp);
  conv(IN_EMB_W, 0, 128, 37, 128, 64, wemb);
  for (int l = 0; l < 3; ++l) {
    int base = IN_GNN0 + 16*l;
    conv(base+12, 0, 128, 128, 128, 128, wq[l]);
    conv(base+4,  0, 128, 128, 128, 128, wk[l]);
    conv(base+14, 0, 128, 128, 128, 128, wv[l]);
    conv(base+10, 0, 128, 128, 128, 128, wo[l]);
    conv(base+0,  0, 256, 128, 256, 128, wf1[l]);
    conv(base+2,  0, 128, 256, 128, 256, wf2[l]);
  }
  conv(IN_POOL_INW, 128*128,   128, 128, 128, 128, wpk);
  conv(IN_POOL_INW, 2*128*128, 128, 128, 128, 128, wpv);
  conv(IN_POOL_OPW, 0, 128, 128, 128, 128, wpo);
  conv(IN_PUR1_W, 0, 128, 512, 128, 512, wpur1);
  conv(IN_PUR2_W, 0, 13, 128, 16, 128, wpur2);
  conv(IN_ATK1_W, 0, 64, 128, 64, 128, watk1);
  conv(IN_ATK2_W, 0, 1, 64, 16, 64, watk2);
  conv(IN_RNF1_W, 0, 64, 128, 64, 128, wrnf1);
  conv(IN_RNF2_W, 0, 1, 64, 16, 64, wrnf2);
  conv(IN_V1_W, 0, 128, 512, 128, 512, wv1);
  conv(IN_V2_W, 0, 64, 128, 64, 128, wv2);
  conv(IN_V3_W, 0, 1, 64, 16, 64, wv3);

  prep_terr<<<(unsigned)(((long)MR*128 + 255)/256), 256, 0, stream>>>(F(IN_OBS), At);
  prep_g<<<(BATCH*32 + 255)/256, 256, 0, stream>>>(F(IN_OBS), Ag);
  mask_k<<<(TPAD*8 + 255)/256, 256, 0, stream>>>((const unsigned char*)d_in[IN_ADJ], mbits);

  // ge = relu(g2(relu(g1(graw))))
  rungemm(stream,4,4, Ag,32,  wg1,32,  F(IN_G1_B),64,  BATCH,64,32, 1, nullptr,0, h1,64,  nullptr);
  rungemm(stream,4,4, h1,64,  wg2,64,  F(IN_G2_B),128, BATCH,128,64,1, ge,128,   nullptr,0, nullptr);
  // c = relu(compress(terr));  e_pre = embed(c)
  rungemm(stream,4,1, At,128, wcmp,128, F(IN_CMP_B),37, MR,48,128, 1, nullptr,0, Cc,64, nullptr);
  rungemm(stream,4,4, Cc,64,  wemb,64,  F(IN_EMB_B),128, MR,128,64, 0, tmp,128, nullptr,0, nullptr);
  int rowsBT = BATCH*TT;
  ln_k<<<(rowsBT+1)/2,64,0,stream>>>(0, tmp, x, xb, F(IN_EMB_LN_G), F(IN_EMB_LN_B), ge, nullptr, rowsBT);

  for (int l = 0; l < 3; ++l) {
    int base = IN_GNN0 + 16*l;
    rungemm(stream,4,4, xb,128, wq[l],128, F(base+13),128, MR,128,128, 0, nullptr,0, nullptr,0, qb);
    rungemm(stream,4,4, xb,128, wk[l],128, F(base+5), 128, MR,128,128, 0, nullptr,0, nullptr,0, kb);
    rungemm(stream,4,4, xb,128, wv[l],128, F(base+15),128, MR,128,128, 0, nullptr,0, nullptr,0, vb);
    attn_k<<<BATCH*4,64,0,stream>>>(qb, kb, vb, mbits, ob);
    rungemm(stream,4,4, ob,128, wo[l],128, F(base+11),128, MR,128,128, 0, tmp,128, nullptr,0, nullptr);
    ln_k<<<(rowsBT+1)/2,64,0,stream>>>(1, tmp, x, xb, F(base+7), F(base+6), nullptr, nullptr, rowsBT);
    rungemm(stream,4,4, xb,128, wf1[l],128, F(base+1),256, MR,256,128, 2, nullptr,0, f1,256, nullptr);
    rungemm(stream,4,4, f1,256, wf2[l],256, F(base+3),128, MR,128,256, 0, tmp,128, nullptr,0, nullptr);
    ln_k<<<(rowsBT+1)/2,64,0,stream>>>(1, tmp, x, xb, F(base+9), F(base+8), nullptr, nullptr, rowsBT);
  }

  // global pool (reuse kb/vb)
  rungemm(stream,4,4, xb,128, wpk,128, F(IN_POOL_INB)+128, 128, MR,128,128, 0, nullptr,0, nullptr,0, kb);
  rungemm(stream,4,4, xb,128, wpv,128, F(IN_POOL_INB)+256, 128, MR,128,128, 0, nullptr,0, nullptr,0, vb);
  qp_k<<<3,128,0,stream>>>(F(IN_POOL_TOK), F(IN_POOL_INW), F(IN_POOL_INB), qp);
  pool_attn<<<BATCH,128,0,stream>>>(qp, kb, vb, att3);
  rungemm(stream,4,4, att3,128, wpo,128, F(IN_POOL_OPB),128, 2048,128,128, 0, proj3,128, nullptr,0, nullptr);
  ln_k<<<(BATCH*3+1)/2,64,0,stream>>>(2, proj3, nullptr, nullptr, F(IN_POOL_LNG), F(IN_POOL_LNB), nullptr, ctx, BATCH*3);
  ctx_build<<<(BATCH*512)/256,256,0,stream>>>(ctx, ge, ctxb);

  // heads
  rungemm(stream,4,4, ctxb,512, wpur1,512, F(IN_PUR1_B),128, BATCH,128,512, 1, nullptr,0, p1,128, nullptr);
  rungemm(stream,4,1, p1,128,   wpur2,128, F(IN_PUR2_B),13,  BATCH,16,128,  3, puro,16, nullptr,0, nullptr);
  rungemm(stream,4,4, xb,128, watk1,128, F(IN_ATK1_B),64, MR,64,128, 1, nullptr,0, a1,64, nullptr);
  rungemm(stream,4,1, a1,64,  watk2,64,  F(IN_ATK2_B),1,  MR,16,64,  3, atko,16, nullptr,0, nullptr);
  rungemm(stream,4,4, xb,128, wrnf1,128, F(IN_RNF1_B),64, MR,64,128, 1, nullptr,0, r1,64, nullptr);
  rungemm(stream,4,1, r1,64,  wrnf2,64,  F(IN_RNF2_B),1,  MR,16,64,  3, rnfo,16, nullptr,0, nullptr);
  rungemm(stream,4,4, ctxb,512, wv1,512, F(IN_V1_B),128, BATCH,128,512, 1, nullptr,0, vh1,128, nullptr);
  rungemm(stream,4,4, vh1,128,  wv2,128, F(IN_V2_B),64,  BATCH,64,128,  1, nullptr,0, vh2,64,  nullptr);
  rungemm(stream,4,1, vh2,64,   wv3,64,  F(IN_V3_B),1,   BATCH,16,64,   0, vo,16, nullptr,0, nullptr);

  long tot_out = (long)BATCH*13 + 2L*BATCH*TT + BATCH;
  gather_out<<<(unsigned)((tot_out + 255)/256),256,0,stream>>>(puro, atko, rnfo, vo, (float*)d_out);
}
